// GATLayer_21586505629926
// MI455X (gfx1250) — matrix operand order repaired and checked by reading, not compiled, not run on hardware
//
#include <hip/hip_runtime.h>

typedef __attribute__((ext_vector_type(16))) _Float16 v16h;
typedef __attribute__((ext_vector_type(8)))  float    v8f;

#define ALPHA_ 0.2f
#define B_   2
#define N_   2048
#define FIN_ 128
#define H_   4
#define D_   32
#define HD_  128          // H_*D_
#define NJT_ (N_/32)      // 64 j-tiles of 32
#define NIT_ (N_/16)      // 128 i-tiles of 16

// K offset pattern for 16-bit WMMA A fragment (per ISA 16x32 layout):
// lane l holds row M=l&15; dword p holds K pair starting at kb + koffA(p),
// kb = 8*(l>>4).  Pattern: {0,2,4,6,16,18,20,22}.
__device__ __forceinline__ int koffA(int p) { return (p < 4) ? 2*p : 2*p + 8; }

// ---------------------------------------------------------------------------
// Prep 1: repack W (128x128 f32, row-major [k][n]) into f16 WMMA B-fragments.
// Fragment (kt,nt): lane l holds col n = nt*16+(l&15); dword p = K pair
// (kb+koffA(p), kb+koffA(p)+1), kb = kt*32 + 8*(l>>4).
// ---------------------------------------------------------------------------
__global__ void pack_w_kernel(const float* __restrict__ W, unsigned* __restrict__ wf) {
    int kt = blockIdx.x >> 3, nt = blockIdx.x & 7;
    int l  = threadIdx.x;
    int n  = nt*16 + (l & 15);
    int kb = kt*32 + ((l >> 4) << 3);
    unsigned* dst = wf + ((size_t)blockIdx.x*32 + l)*8;
#pragma unroll
    for (int p = 0; p < 8; ++p) {
        int k0 = kb + koffA(p);
        union { _Float16 h[2]; unsigned u; } c;
        c.h[0] = (_Float16)W[(size_t)k0*HD_ + n];
        c.h[1] = (_Float16)W[(size_t)(k0+1)*HD_ + n];
        dst[p] = c.u;
    }
}

// ---------------------------------------------------------------------------
// Prep 2: u[sd][h][f] = sum_d W[f][h*32+d] * att[h][sd*32+d]   (exact f32)
// e = (x@W)@a  ==  x@(W@a) by associativity -> tiny GEMV instead of reducing
// WMMA fragments across lanes.
// ---------------------------------------------------------------------------
__global__ void compute_u_kernel(const float* __restrict__ W,
                                 const float* __restrict__ att,
                                 float* __restrict__ u) {
    int t = blockIdx.x*blockDim.x + threadIdx.x;   // 1024 threads
    int f = t & 127, h = (t >> 7) & 3, sd = t >> 9;
    float acc = 0.f;
#pragma unroll 8
    for (int d = 0; d < D_; ++d)
        acc += W[(size_t)f*HD_ + h*D_ + d] * att[h*2*D_ + sd*D_ + d];
    u[((size_t)sd*H_ + h)*FIN_ + f] = acc;
}

// ---------------------------------------------------------------------------
// Prep 3: e[sd][b][h][n] = dot(x[b][n][:], u[sd][h][:])   (128-length, f32)
// ---------------------------------------------------------------------------
__global__ void compute_e_kernel(const float* __restrict__ x,
                                 const float* __restrict__ u,
                                 float* __restrict__ e) {
    int t = blockIdx.x*blockDim.x + threadIdx.x;   // 32768 threads
    int idx8 = t & 7, sd = idx8 >> 2, h = idx8 & 3;
    int rg = t >> 3;                               // flat b*N+n
    const float* xr = x + (size_t)rg*FIN_;
    const float* ur = u + ((size_t)sd*H_ + h)*FIN_;
    float acc = 0.f;
#pragma unroll 8
    for (int f = 0; f < FIN_; ++f) acc += xr[f]*ur[f];
    int b = rg / N_, n = rg % N_;
    e[(((size_t)sd*B_ + b)*H_ + h)*N_ + n] = acc;
}

// ---------------------------------------------------------------------------
// Prep 4: adj (int32 0/1) -> 1 bit. 32MB -> 1MB of mask traffic.
// ---------------------------------------------------------------------------
__global__ void pack_adj_kernel(const int* __restrict__ adj, unsigned* __restrict__ bits) {
    int t = blockIdx.x*blockDim.x + threadIdx.x;   // B*N*N threads
    unsigned long long m = __ballot(adj[t] != 0);
    if ((threadIdx.x & 31) == 0) bits[t >> 5] = (unsigned)m;
}

// ---------------------------------------------------------------------------
// Stage 1: x_proj = x@W via WMMA; emit f16 x_proj directly in B-fragment
// order for the attention PV matmul.  64 threads (2 waves): each wave does a
// 16-row tile of x (8 col-tiles x 4 K-steps of wmma), stages f16 through LDS,
// then the block repacks its 32 rows into 8 (h,dhalf) B-fragments.
// vf layout: [b][h][jt][dhalf][lane][dword*8]
// ---------------------------------------------------------------------------
__global__ void proj_pack_v_kernel(const float* __restrict__ x,
                                   const unsigned* __restrict__ wf,
                                   unsigned* __restrict__ vf) {
    __shared__ _Float16 xp[32][HD_];               // 8 KB
    int l  = threadIdx.x & 31;
    int w  = threadIdx.x >> 5;
    int mi = l & 15, hi = l >> 4;
    int rg0 = blockIdx.x * 32;                     // block's first flat row
    int row = rg0 + w*16 + mi;                     // A-fragment row (M = l&15)

    // Load A fragments for 4 K-tiles (K pairs are contiguous in x's F dim).
    v16h a[4];
#pragma unroll
    for (int kt = 0; kt < 4; ++kt) {
#pragma unroll
        for (int p = 0; p < 8; ++p) {
            int k0 = kt*32 + hi*8 + koffA(p);
            float2 xv = *(const float2*)(x + (size_t)row*FIN_ + k0);
            a[kt][2*p]   = (_Float16)xv.x;
            a[kt][2*p+1] = (_Float16)xv.y;
        }
    }
    // 8 output col-tiles of 16 (h*32+dhalf*16), each K=128 in 4 wmma steps.
#pragma unroll
    for (int nt = 0; nt < 8; ++nt) {
        v8f c = {};
#pragma unroll
        for (int kt = 0; kt < 4; ++kt) {
            v16h bfr = *(const v16h*)(wf + (size_t)((kt*8 + nt)*32 + l)*8);
            c = __builtin_amdgcn_wmma_f32_16x16x32_f16(false, a[kt], false, bfr,
                                                       (short)0, c, false, false);
        }
#pragma unroll
        for (int r = 0; r < 8; ++r)                 // C layout: M = r + 8*hi
            xp[w*16 + r + 8*hi][nt*16 + mi] = (_Float16)c[r];
    }
    __syncthreads();

    // Repack 32 rows x 128 cols into 8 B-fragments (K=32 rows, N=16 d-cols).
    int b  = rg0 / N_;
    int jt = (rg0 % N_) >> 5;
#pragma unroll
    for (int f = 0; f < 4; ++f) {
        int fr = w*4 + f;                           // 0..7 -> (h, dhalf)
        int h = fr >> 1, dh = fr & 1;
        int col = h*D_ + dh*16 + mi;                // B lane = d-col
        int kb2 = hi << 3;                          // K pairs koffA(p) (+8 upper half)
        unsigned* dst = vf + (size_t)((((b*H_ + h)*NJT_ + jt)*2 + dh)*32 + l)*8;
#pragma unroll
        for (int p = 0; p < 8; ++p) {
            int n0 = kb2 + koffA(p);
            union { _Float16 hh[2]; unsigned u; } cc;
            cc.hh[0] = xp[n0][col];
            cc.hh[1] = xp[n0+1][col];
            dst[p] = cc.u;
        }
    }
}

// ---------------------------------------------------------------------------
// Stage 2: fused logits + LeakyReLU + bitmask + online softmax + PV wmma.
// One wave per (b, h, 16-row i-tile).  Logits are computed per-lane at the
// exact A-fragment positions, so P feeds WMMA with zero cross-lane repack.
// Row stats live duplicated in lanes l and l^16 (combined via shfl_xor 16).
// ---------------------------------------------------------------------------
__global__ void gat_attn_kernel(const float* __restrict__ e,
                                const unsigned* __restrict__ bits,
                                const unsigned* __restrict__ vf,
                                float* __restrict__ out) {
    int bid = blockIdx.x;
    int it = bid & (NIT_ - 1);
    int h  = (bid >> 7) & (H_ - 1);
    int b  = bid >> 9;
    int l  = threadIdx.x;
    int mi = l & 15, hi = l >> 4, kb = hi*8;
    int ig = it*16 + mi;

    float es = e[((size_t)b*H_ + h)*N_ + ig];                       // sd=0 (src)
    const float* ed = e + (((size_t)B_ + b)*H_ + h)*N_;             // sd=1 (dst)
    const unsigned* rowbits = bits + ((size_t)b*N_ + ig)*NJT_;
    const unsigned* vbase   = vf + (size_t)((b*H_ + h)*NJT_)*512;   // 2*32*8 dwords/jt

    float m = -__builtin_inff(), s = 0.f;
    v8f acc0 = {}, acc1 = {};

    for (int jt = 0; jt < NJT_; ++jt) {
        unsigned aw = rowbits[jt];                  // 32 adjacency bits, one load
        int jbase = jt*32;
        float lg[16];
        float tmax = -__builtin_inff();
#pragma unroll
        for (int p = 0; p < 8; ++p) {
#pragma unroll
            for (int q = 0; q < 2; ++q) {
                int jj = kb + koffA(p) + q;         // bit index == local j
                float v = es + ed[jbase + jj];
                v = (v >= 0.f) ? v : ALPHA_*v;      // LeakyReLU
                v = ((aw >> jj) & 1u) ? v : -1e30f; // mask
                lg[2*p + q] = v;
                tmax = fmaxf(tmax, v);
            }
        }
        tmax = fmaxf(tmax, __shfl_xor(tmax, 16));   // full row tile-max
        float nm    = fmaxf(m, tmax);
        float scale = __expf(m - nm);               // first iter: exp(-inf)=0
        float ssum  = 0.f;
        v16h af;
#pragma unroll
        for (int k = 0; k < 16; ++k) {
            float pv = __expf(lg[k] - nm);
            ssum += pv;
            af[k] = (_Float16)pv;
        }
        ssum += __shfl_xor(ssum, 16);
        s = s*scale + ssum;
        m = nm;
        // Rescale accumulators; factor for row M=r+8*hi lives in lane r+8*hi.
#pragma unroll
        for (int r = 0; r < 8; ++r) {
            float sc = __shfl(scale, r + 8*hi);
            acc0[r] *= sc;
            acc1[r] *= sc;
        }
        const v16h* vb = (const v16h*)(vbase + (size_t)jt*512);
        v16h vb0 = vb[l];
        v16h vb1 = vb[32 + l];
        acc0 = __builtin_amdgcn_wmma_f32_16x16x32_f16(false, af, false, vb0,
                                                      (short)0, acc0, false, false);
        acc1 = __builtin_amdgcn_wmma_f32_16x16x32_f16(false, af, false, vb1,
                                                      (short)0, acc1, false, false);
    }
    // Normalize and store: out[b][row][h*32 + dhalf*16 + mi]
#pragma unroll
    for (int r = 0; r < 8; ++r) {
        float inv = 1.0f / __shfl(s, r + 8*hi);
        int rowg = it*16 + r + 8*hi;
        size_t o = ((size_t)b*N_ + rowg)*HD_ + h*D_ + mi;
        out[o]      = acc0[r]*inv;
        out[o + 16] = acc1[r]*inv;
    }
}

// ---------------------------------------------------------------------------
// Workspace layout (bytes):
//   wf   @ 0        : 32768    (W fragments, f16 pairs)
//   u    @ 32768    : 4096     (W@a, f32)
//   e    @ 36864    : 131072   (e_src/e_dst, f32)
//   bits @ 167936   : 1048576  (adjacency bitmask)
//   vf   @ 1216512  : 1048576  (x_proj B-fragments, f16 pairs)
//   total ~2.17 MB
// ---------------------------------------------------------------------------
extern "C" void kernel_launch(void* const* d_in, const int* in_sizes, int n_in,
                              void* d_out, int out_size, void* d_ws, size_t ws_size,
                              hipStream_t stream) {
    const float* x   = (const float*)d_in[0];
    const int*   adj = (const int*)  d_in[1];
    const float* W   = (const float*)d_in[2];
    const float* att = (const float*)d_in[3];
    float* out = (float*)d_out;

    char* ws = (char*)d_ws;
    unsigned* wf   = (unsigned*)(ws + 0);
    float*    u    = (float*)   (ws + 32768);
    float*    e    = (float*)   (ws + 36864);
    unsigned* bits = (unsigned*)(ws + 167936);
    unsigned* vf   = (unsigned*)(ws + 1216512);

    pack_w_kernel     <<<32, 32, 0, stream>>>(W, wf);
    compute_u_kernel  <<<4, 256, 0, stream>>>(W, att, u);
    compute_e_kernel  <<<128, 256, 0, stream>>>(x, u, e);
    pack_adj_kernel   <<<(B_*N_*N_)/256, 256, 0, stream>>>(adj, bits);
    proj_pack_v_kernel<<<(B_*N_)/32, 64, 0, stream>>>(x, wf, vf);
    gat_attn_kernel   <<<B_*H_*NIT_, 32, 0, stream>>>(e, bits, vf, out);
}
